// MyLSTM_85383949844507
// MI455X (gfx1250) — matrix operand order repaired and checked by reading, not compiled, not run on hardware
//
#include <hip/hip_runtime.h>
#include <math.h>

typedef __attribute__((ext_vector_type(16))) _Float16 v16h;
typedef __attribute__((ext_vector_type(8)))  _Float16 v8h;
typedef __attribute__((ext_vector_type(8)))  float    v8f;
typedef int vs4i __attribute__((vector_size(16)));    // async builtin pointee type

#define EMB   256
#define HID   256
#define BATCH 128
#define SEQ   1024
#define KT_N  16                              // K chunks of 32 (K total = 512)
#define HT_N  16                              // hidden 16-col tiles
#define MT_N  8                               // batch 16-row tiles
#define GATE_STRIDE (KT_N * HT_N * 32 * 16)   // halves per gate in wp (131072)

// LDS: weights 64KB + A fragments 64KB, each buffer split into two 16B planes
// (lo = halves 0..7, hi = halves 8..15) at lane-stride 16B -> conflict-free
// ds_load_b128 (all 64 banks covered exactly twice per wave request).
#define WL_HALVES (4 * KT_N * 32 * 16)        // 32768
#define AL_HALVES (4 * KT_N * 32 * 16)        // 32768
#define PLANE     (WL_HALVES / 2)             // 16384 halves per plane
#define WL_GATE8  (KT_N * 32 * 8)             // 4096 halves: per-gate stride in a plane
#define LDS_BYTES ((WL_HALVES + AL_HALVES) * 2)   // 131072 B

// ---------------------------------------------------------------------------
// Pre-swizzle the four f32 weight matrices [512,256] into f16 B-fragments.
// wp[(((g*16 + kt)*16 + ht)*32 + lane)*16 + j]
//   B 32x16 fragment (ISA 7.12.2): lane<16 -> col=lane,    K = kk + (j&7) + 16*(j>>3)
//                                  lane>=16 -> col=lane-16, K = kk + 8 + (j&7) + 16*(j>>3)
// ---------------------------------------------------------------------------
__global__ void prep_weights(const float* __restrict__ Wf, const float* __restrict__ Wi,
                             const float* __restrict__ Wc, const float* __restrict__ Wo,
                             _Float16* __restrict__ wp) {
    int tid  = blockIdx.x * blockDim.x + threadIdx.x;   // 0 .. 4*GATE_STRIDE-1
    int j    = tid & 15;
    int lane = (tid >> 4) & 31;
    int ht   = (tid >> 9) & 15;
    int kt   = (tid >> 13) & 15;
    int g    = tid >> 17;
    int k = kt * 32 + ((lane >> 4) << 3) + (j & 7) + ((j >> 3) << 4);   // 0..511
    int n = ht * 16 + (lane & 15);                      // 0..255
    const float* W = (g == 0) ? Wf : (g == 1) ? Wi : (g == 2) ? Wc : Wo;
    wp[tid] = (_Float16)W[k * HID + n];
}

__global__ void init_state(float* __restrict__ a0, float* __restrict__ c0) {
    int i = blockIdx.x * blockDim.x + threadIdx.x;
    a0[i] = 0.0f;
    c0[i] = 0.0f;
}

__device__ inline v8h pack8v(float4 u0, float4 u1) {
    v8h r;
    r[0] = (_Float16)u0.x; r[1] = (_Float16)u0.y;
    r[2] = (_Float16)u0.z; r[3] = (_Float16)u0.w;
    r[4] = (_Float16)u1.x; r[5] = (_Float16)u1.y;
    r[6] = (_Float16)u1.z; r[7] = (_Float16)u1.w;
    return r;
}

__device__ inline v16h cat16(v8h lo, v8h hi) {
    return __builtin_shufflevector(lo, hi, 0, 1, 2, 3, 4, 5, 6, 7,
                                           8, 9, 10, 11, 12, 13, 14, 15);
}

// ---------------------------------------------------------------------------
// One recurrent step. Grid (2,16), block 128 (4 waves).
// Block owns (mt group of 4) x (one hidden tile ht=blockIdx.y); wave w owns
// batch tile mt = bx*4+w. All waves share the B fragments staged in LDS via
// async global->LDS copies; A fragments staged cooperatively as f16.
// ---------------------------------------------------------------------------
__global__ __launch_bounds__(128)
void lstm_step(const float*    __restrict__ x,
               const _Float16* __restrict__ wp,
               const float* __restrict__ bfp, const float* __restrict__ bip,
               const float* __restrict__ bcp, const float* __restrict__ bop,
               const float* __restrict__ a_prev, const float* __restrict__ c_prev,
               float* __restrict__ a_next, float* __restrict__ c_next,
               float* __restrict__ out, int t) {
    extern __shared__ uint4 smem4[];
    _Float16* wl = (_Float16*)smem4;          // planes: [2][g][kt][lane][8]
    _Float16* al = wl + WL_HALVES;            // planes: [2][wm][kt][lane][8]

    const int tid  = threadIdx.x;
    const int lane = tid & 31;
    const int w    = tid >> 5;                // wave id = local mt
    const int bx   = blockIdx.x;              // 0..1   (mt group)
    const int by   = blockIdx.y;              // 0..15  (ht)
    const int mt   = bx * 4 + w;

    // ---- stage weight slice (64KB) via async global->LDS b128 copies ----
    for (int c = tid; c < WL_HALVES / 8; c += 128) {   // 4096 16-byte chunks
        const int j8   = c & 1;                        // plane select
        const int rest = c >> 1;                       // (g*16+kt)*32 + ln
        const int ln = rest & 31;
        const int kt = (rest >> 5) & 15;
        const int g  = rest >> 9;
        const _Float16* src =
            wp + (((size_t)(g * 16 + kt) * 16 + by) * 32 + ln) * 16 + j8 * 8;
        _Float16* dst = wl + (size_t)j8 * PLANE + (size_t)rest * 8;
#if __has_builtin(__builtin_amdgcn_global_load_async_to_lds_b128)
        __builtin_amdgcn_global_load_async_to_lds_b128(
            (__attribute__((address_space(1))) vs4i*)const_cast<_Float16*>(src),
            (__attribute__((address_space(3))) vs4i*)dst, 0, 0);
#else
        *(uint4*)dst = *(const uint4*)src;
#endif
    }

    // ---- stage A fragments: h_t = [x_t ; a_prev] rows for 4 batch tiles ----
    // A 16x32 f16 fragment (ISA 7.12.2): row = lane&15, K-offset +8 for
    // lanes>=16, halves j<8 -> K+0..7, j>=8 -> K+16..23.
    for (int p = tid; p < AL_HALVES / 16; p += 128) {  // 2048 fragments
        const int ln = p & 31;
        const int kt = (p >> 5) & 15;
        const int wm = p >> 9;
        const int mg = (bx * 4 + wm) * 16 + (ln & 15);
        const int koff = (ln >> 4) << 3;
        const int kk = kt * 32;
        const float* src = (kk < EMB)
            ? (x + ((size_t)mg * SEQ + (size_t)t) * EMB + kk + koff)
            : (a_prev + (size_t)mg * HID + (kk - EMB) + koff);
        float4 r0 = ((const float4*)src)[0];
        float4 r1 = ((const float4*)src)[1];
        float4 r2 = ((const float4*)(src + 16))[0];
        float4 r3 = ((const float4*)(src + 16))[1];
        *(v8h*)(al + (size_t)p * 8)         = pack8v(r0, r1);   // lo plane
        *(v8h*)(al + PLANE + (size_t)p * 8) = pack8v(r2, r3);   // hi plane
        // prefetch next timestep's x slice onto the next launch's critical path
        if (kk < EMB && t + 1 < SEQ)
            __builtin_prefetch(src + EMB, 0, 1);
    }

#if __has_builtin(__builtin_amdgcn_s_wait_asynccnt)
    __builtin_amdgcn_s_wait_asynccnt(0);
#else
    asm volatile("s_wait_asynccnt 0" ::: "memory");
#endif
    __syncthreads();

    // ---- hot loop: pure LDS -> WMMA, 4 gates share one A fragment ----
    v8f accF = {}, accI = {}, accC = {}, accO = {};
    const _Float16* abase = al + (size_t)w * WL_GATE8;   // this wave's mt tile
    for (int kt = 0; kt < KT_N; ++kt) {
        const size_t fo = ((size_t)kt * 32 + lane) * 8;
        v16h a  = cat16(*(const v8h*)(abase + fo),
                        *(const v8h*)(abase + fo + PLANE));
        const _Float16* wb = wl + fo;
        v16h bF = cat16(*(const v8h*)(wb),
                        *(const v8h*)(wb + PLANE));
        v16h bI = cat16(*(const v8h*)(wb + WL_GATE8),
                        *(const v8h*)(wb + WL_GATE8 + PLANE));
        v16h bC = cat16(*(const v8h*)(wb + 2 * WL_GATE8),
                        *(const v8h*)(wb + 2 * WL_GATE8 + PLANE));
        v16h bO = cat16(*(const v8h*)(wb + 3 * WL_GATE8),
                        *(const v8h*)(wb + 3 * WL_GATE8 + PLANE));
        accF = __builtin_amdgcn_wmma_f32_16x16x32_f16(false, a, false, bF, (short)0, accF, false, false);
        accI = __builtin_amdgcn_wmma_f32_16x16x32_f16(false, a, false, bI, (short)0, accI, false, false);
        accC = __builtin_amdgcn_wmma_f32_16x16x32_f16(false, a, false, bC, (short)0, accC, false, false);
        accO = __builtin_amdgcn_wmma_f32_16x16x32_f16(false, a, false, bO, (short)0, accO, false, false);
    }

    // ---- fused gate math; C/D layout: acc[r] -> (M=r+8*(lane>=16), N=lane&15)
    const int n     = by * 16 + (lane & 15);
    const int mbase = mt * 16 + ((lane >> 4) << 3);
    const float vbf = bfp[n], vbi = bip[n], vbc = bcp[n], vbo = bop[n];

    #pragma unroll
    for (int r = 0; r < 8; ++r) {
        const int m = mbase + r;
        const size_t sidx = (size_t)m * HID + n;
        float fg = fmaxf(accF[r] + vbf, 0.0f);     // ReLU gates (faithful to ref)
        float ig = fmaxf(accI[r] + vbi, 0.0f);
        float cc = tanhf(accC[r] + vbc);
        float og = fmaxf(accO[r] + vbo, 0.0f);
        float cp = c_prev[sidx];
        float cn = fg * cp + ig * cc;
        float an = tanhf(cn) * og;
        c_next[sidx] = cn;
        a_next[sidx] = an;
        out[((size_t)m * SEQ + (size_t)t) * HID + n] = an;
    }
}

__global__ void finalize(const float* __restrict__ a_last, const float* __restrict__ c_last,
                         float* __restrict__ out_tail) {
    int i = blockIdx.x * blockDim.x + threadIdx.x;
    out_tail[i] = a_last[i];
    out_tail[BATCH * HID + i] = c_last[i];
}

// ---------------------------------------------------------------------------
extern "C" void kernel_launch(void* const* d_in, const int* in_sizes, int n_in,
                              void* d_out, int out_size, void* d_ws, size_t ws_size,
                              hipStream_t stream) {
    const float* x  = (const float*)d_in[0];
    const float* Wf = (const float*)d_in[1];
    const float* bf = (const float*)d_in[2];
    const float* Wi = (const float*)d_in[3];
    const float* bi = (const float*)d_in[4];
    const float* Wc = (const float*)d_in[5];
    const float* bc = (const float*)d_in[6];
    const float* Wo = (const float*)d_in[7];
    const float* bo = (const float*)d_in[8];
    float* out = (float*)d_out;

    const size_t BH = (size_t)BATCH * HID;       // 32768
    float* a0 = (float*)d_ws;
    float* a1 = a0 + BH;
    float* c0 = a1 + BH;
    float* c1 = c0 + BH;
    _Float16* wp = (_Float16*)(c1 + BH);         // 4*GATE_STRIDE halves = 1 MB

    prep_weights<<<(4 * GATE_STRIDE) / 256, 256, 0, stream>>>(Wf, Wi, Wc, Wo, wp);
    init_state<<<(int)(BH / 256), 256, 0, stream>>>(a0, c0);

    float* ab[2] = {a0, a1};
    float* cb[2] = {c0, c1};
    for (int t = 0; t < SEQ; ++t) {
        const int p = t & 1;
        lstm_step<<<dim3(MT_N / 4, HT_N), 128, LDS_BYTES, stream>>>(
            x, wp, bf, bi, bc, bo,
            ab[p], cb[p], ab[p ^ 1], cb[p ^ 1], out, t);
    }
    // After 1024 steps the final state lives in buffer index (1024 & 1) == 0.
    finalize<<<(int)(BH / 256), 256, 0, stream>>>(ab[0], cb[0],
                                                  out + (size_t)BATCH * SEQ * HID);
}
